// TextDiffusionTransformer_18433999634976
// MI455X (gfx1250) — matrix operand order repaired and checked by reading, not compiled, not run on hardware
//
#include <hip/hip_runtime.h>
#include <math.h>

typedef _Float16 f16_t;
typedef __attribute__((ext_vector_type(16))) _Float16 v16h;
typedef __attribute__((ext_vector_type(8)))  _Float16 v8h;
typedef __attribute__((ext_vector_type(4)))  _Float16 v4h;
typedef __attribute__((ext_vector_type(8)))  float    v8f;
typedef __attribute__((ext_vector_type(4)))  float    v4f;
typedef __attribute__((ext_vector_type(2)))  float    v2f;

#define BB      2
#define CTXN    1800
#define DIFFN   200
#define DMODEL  512
#define NHEAD   32
#define VOCAB   5000
#define NLAYER  12
#define WWIN    250
#define SEQ     2000
#define FFD     2048
#define HDIM    16

// LDS row stride (in halves): 40 halves = 80 bytes keeps the fragment chunks
// at half-offsets {0,8,16,24,32} 16-byte aligned for ds_load_b128.
#define LDSTR   40
#define TILESZ  (64 * LDSTR)              // halves per LDS tile buffer

// ---------------------------------------------------------------------------
// GEMM: C[M,N] = act(A[M,K] @ W[K,N] + bias[N]); f32 in/out, f16 WMMA compute.
// Block = 128 threads (4 waves in 2x2), block tile 64x64, K stepped by 32.
// Each wave owns a 32x32 register tile (2x2 accumulators -> 4 WMMA / K-step).
// Software-pipelined with DOUBLE-BUFFERED LDS: tile t+1's 8 global b128 loads
// are issued before the WMMA work on tile t, and committed to the other LDS
// buffer afterwards; one barrier per K-step.  Staging is branch-free (needs
// K%4==0 && N%4==0, true for all call sites): clamped b128 load + one select
// per float4.  A staged [m][k]; W staged transposed [n][k] via per-thread 4x4
// register transpose, so every WMMA fragment is two contiguous ds_load_b128.
// act: 0 = none, 1 = exact GELU.
// ---------------------------------------------------------------------------
__global__ __launch_bounds__(128)
void gemm_f16_kernel(const float* __restrict__ A, const float* __restrict__ W,
                     const float* __restrict__ bias, float* __restrict__ C,
                     int M, int K, int N, int act)
{
    __shared__ __align__(16) f16_t As[2 * TILESZ];   // [buf][m 0..63][k 0..31]
    __shared__ __align__(16) f16_t Bt[2 * TILESZ];   // [buf][n 0..63][k 0..31]

    const int tid  = threadIdx.x;
    const int lane = tid & 31;
    const int wave = tid >> 5;            // 0..3
    const int wm   = wave >> 1;           // 0..1
    const int wn   = wave & 1;            // 0..1
    const int half = lane >> 4;           // 0..1
    const int l16  = lane & 15;           // 0..15
    const int m0   = blockIdx.y * 64;
    const int n0   = blockIdx.x * 64;

    const v4f vzero = {0.f, 0.f, 0.f, 0.f};
    v8f acc[2][2] = {{{}, {}}, {{}, {}}};

    // A staging map: float4 f = tid + 128*i -> row r = (tid>>3)+16*i, col4 tid&7
    const int ar  = tid >> 3;
    const int ac4 = tid & 7;
    // B staging map: per-thread 4x4 block: n-block nb=tid&15, k-block kb=tid>>4
    const int nb  = tid & 15;
    const int kb  = tid >> 4;             // 0..7

    // ---- branch-free tile loaders (issue 8 b128 loads, no waits between) ----
    auto load_tile = [&](int k0, v4f (&av)[4], v4f (&bv)[4]) {
        const int gc   = k0 + (ac4 << 2);
        const bool kokA = gc < K;                     // K%4==0 -> whole float4
        const int gcc  = kokA ? gc : K - 4;
        #pragma unroll
        for (int i = 0; i < 4; ++i) {
            const int r   = ar + (i << 4);            // 0..63
            const int gr  = m0 + r;
            const bool mok = gr < M;
            const int grc = mok ? gr : M - 1;
            v4f fv = *(const v4f*)(A + (size_t)grc * K + gcc);
            av[i] = (mok & kokA) ? fv : vzero;
        }
        const int gn   = n0 + (nb << 2);
        const bool nok = gn < N;                      // N%4==0 -> whole float4
        const int gnc  = nok ? gn : N - 4;
        #pragma unroll
        for (int rr = 0; rr < 4; ++rr) {
            const int gk  = k0 + (kb << 2) + rr;
            const bool kok = gk < K;
            const int gkc = kok ? gk : K - 1;
            v4f fv = *(const v4f*)(W + (size_t)gkc * N + gnc);
            bv[rr] = (nok & kok) ? fv : vzero;
        }
    };
    // ---- convert + store a staged tile into LDS buffer `buf` ----
    auto commit_tile = [&](const v4f (&av)[4], const v4f (&bv)[4], int buf) {
        f16_t* Ab = As + buf * TILESZ;
        f16_t* Bb = Bt + buf * TILESZ;
        #pragma unroll
        for (int i = 0; i < 4; ++i)
            *(v4h*)&Ab[(ar + (i << 4)) * LDSTR + (ac4 << 2)] =
                __builtin_convertvector(av[i], v4h);
        #pragma unroll
        for (int e = 0; e < 4; ++e) {
            v4h cv = { (f16_t)bv[0][e], (f16_t)bv[1][e],
                       (f16_t)bv[2][e], (f16_t)bv[3][e] };
            *(v4h*)&Bb[((nb << 2) + e) * LDSTR + (kb << 2)] = cv;
        }
    };

    // prologue: stage tile 0 into buffer 0
    {
        v4f a0[4], b0[4];
        load_tile(0, a0, b0);
        commit_tile(a0, b0, 0);
    }

    int t = 0;
    for (int k0 = 0; k0 < K; k0 += 32, ++t) {
        const int buf   = t & 1;
        const bool have = (k0 + 32) < K;

        // issue next tile's global loads before touching LDS (overlaps WMMA)
        v4f nxtA[4], nxtB[4];
        if (have) load_tile(k0 + 32, nxtA, nxtB);

        // prefetch tile t+2 (global_prefetch_b8)
        if (k0 + 64 < K) {
            int gr = m0 + (tid >> 1); if (gr >= M) gr = M - 1;
            __builtin_prefetch(&A[(size_t)gr * K + k0 + 64 + ((tid & 1) << 4)], 0, 0);
            int kr = k0 + 64 + (tid & 31); if (kr >= K) kr = K - 1;
            __builtin_prefetch(&W[(size_t)kr * N + n0 + ((tid >> 5) << 4)], 0, 0);
        }

        __syncthreads();   // commits of tile t (buffer `buf`) now visible

        // ---- fragments: two 16B LDS loads each (ISA 7.12.2 layouts) ----
        const f16_t* Ab = As + buf * TILESZ;
        const f16_t* Bb = Bt + buf * TILESZ;
        v16h af[2], bf[2];
        #pragma unroll
        for (int s = 0; s < 2; ++s) {
            // A 16x32: lane half h needs K = {8h..8h+7} U {8h+16..8h+23}
            const f16_t* arp = &Ab[(wm * 32 + s * 16 + l16) * LDSTR + half * 8];
            v8h alo = *(const v8h*)(arp);
            v8h ahi = *(const v8h*)(arp + 16);
            af[s] = __builtin_shufflevector(alo, ahi, 0, 1, 2, 3, 4, 5, 6, 7,
                                            8, 9, 10, 11, 12, 13, 14, 15);
            // B 32x16: lane half h needs K = {8h..8h+7} U {8h+16..8h+23}
            const f16_t* brp = &Bb[(wn * 32 + s * 16 + l16) * LDSTR + half * 8];
            v8h blo = *(const v8h*)(brp);
            v8h bhi = *(const v8h*)(brp + 16);
            bf[s] = __builtin_shufflevector(blo, bhi, 0, 1, 2, 3, 4, 5, 6, 7,
                                            8, 9, 10, 11, 12, 13, 14, 15);
        }

        #pragma unroll
        for (int i = 0; i < 2; ++i)
            #pragma unroll
            for (int j = 0; j < 2; ++j)
                acc[i][j] = __builtin_amdgcn_wmma_f32_16x16x32_f16(
                    false, af[i], false, bf[j], (short)0, acc[i][j], false, false);

        // commit tile t+1 into the other buffer; safe with one barrier since
        // buffer reuse distance is 2 iterations.
        if (have) commit_tile(nxtA, nxtB, buf ^ 1);
    }

    // ---- epilogue: C frag -> lane = N%16, VGPR r -> M = r + 8*half ----
    #pragma unroll
    for (int i = 0; i < 2; ++i) {
        #pragma unroll
        for (int j = 0; j < 2; ++j) {
            const int col = n0 + wn * 32 + j * 16 + l16;
            const bool colok = col < N;
            #pragma unroll
            for (int r = 0; r < 8; ++r) {
                int row = m0 + wm * 32 + i * 16 + r + (half << 3);
                if (colok && row < M) {
                    float vo = acc[i][j][r] + bias[col];
                    if (act == 1)
                        vo = 0.5f * vo * (1.f + erff(vo * 0.70710678118654752440f));
                    C[(size_t)row * N + col] = vo;
                }
            }
        }
    }
}

// ---------------------------------------------------------------------------
// Row LayerNorm over D=512 (256 threads, 2 elems/thread); optional residual:
//   x[row] = LN(x[row] + resid[row]) * g + b
// ---------------------------------------------------------------------------
__global__ __launch_bounds__(256)
void ln_kernel(float* __restrict__ x, const float* __restrict__ resid,
               const float* __restrict__ g, const float* __restrict__ b,
               float eps)
{
    __shared__ float red[256];
    const int row = blockIdx.x;
    const int t   = threadIdx.x;
    float* xr = x + (size_t)row * DMODEL;

    float v0 = xr[t], v1 = xr[t + 256];
    if (resid) {
        const float* rr = resid + (size_t)row * DMODEL;
        v0 += rr[t]; v1 += rr[t + 256];
    }
    red[t] = v0 + v1;
    __syncthreads();
    for (int st = 128; st > 0; st >>= 1) {
        if (t < st) red[t] += red[t + st];
        __syncthreads();
    }
    float mean = red[0] * (1.f / DMODEL);
    __syncthreads();

    float d0 = v0 - mean, d1 = v1 - mean;
    red[t] = d0 * d0 + d1 * d1;
    __syncthreads();
    for (int st = 128; st > 0; st >>= 1) {
        if (t < st) red[t] += red[t + st];
        __syncthreads();
    }
    float inv = rsqrtf(red[0] * (1.f / DMODEL) + eps);

    xr[t]       = d0 * inv * g[t]       + b[t];
    xr[t + 256] = d1 * inv * g[t + 256] + b[t + 256];
}

// ---------------------------------------------------------------------------
// Sinusoidal timestep features: tf[b, 0:256]=cos(t*f_i), tf[b, 256:512]=sin
// ---------------------------------------------------------------------------
__global__ void tfreq_kernel(const float* __restrict__ t, float* __restrict__ tf)
{
    int b = blockIdx.x;
    int i = threadIdx.x;                               // 0..255
    float f = __expf(-9.210340371976184f * (float)i * (1.f / 256.f));
    float a = t[b] * f;
    tf[(size_t)b * DMODEL + i]       = cosf(a);
    tf[(size_t)b * DMODEL + 256 + i] = sinf(a);
}

// ---------------------------------------------------------------------------
// Assemble pre-LN embeddings:
//   x[b,s,:] = (s<CTX ? ctx_emb[ids] : dproj_out) + pos_emb[s] + t_emb[b]
//              + lf_pos[2+s] + lf_tok
// ---------------------------------------------------------------------------
__global__ __launch_bounds__(256)
void build_x_kernel(const int* __restrict__ ids, const float* __restrict__ ctx_emb,
                    const float* __restrict__ dbuf, const float* __restrict__ pos_emb,
                    const float* __restrict__ temb, const float* __restrict__ lf_pos,
                    const float* __restrict__ lf_tok, float* __restrict__ x)
{
    const int row = blockIdx.x;                 // 0 .. B*S-1
    const int b   = row / SEQ;
    const int s   = row % SEQ;
    #pragma unroll
    for (int e = 0; e < 2; ++e) {
        int d = threadIdx.x + (e << 8);
        float base;
        if (s < CTXN)
            base = ctx_emb[(size_t)ids[b * CTXN + s] * DMODEL + d];
        else
            base = dbuf[(size_t)(b * DIFFN + (s - CTXN)) * DMODEL + d];
        x[(size_t)row * DMODEL + d] =
            base + pos_emb[(size_t)s * DMODEL + d] + temb[(size_t)b * DMODEL + d] +
            lf_pos[(size_t)(2 + s) * DMODEL + d] + lf_tok[d];
    }
}

// ---------------------------------------------------------------------------
// Sliding-window attention (one-sided window W=250) with f32 WMMA.
// One wave per 16-query tile per (b,h). q/k/v/o layout: [(b*S+s)*512 + h*16+hd].
// Flash-style online softmax over 33 key tiles of 16 covering [q0-256, q0+272).
// ---------------------------------------------------------------------------
__global__ __launch_bounds__(32)
void band_attn_kernel(const float* __restrict__ q, const float* __restrict__ k,
                      const float* __restrict__ v, float* __restrict__ o)
{
    __shared__ float pT[16 * 17];

    const int lane = threadIdx.x;
    const int half = lane >> 4;
    const int c16  = lane & 15;
    const int q0   = blockIdx.x << 4;            // query tile start (0..1984)
    const int bh   = blockIdx.y;                 // 0..63
    const int b    = bh >> 5;
    const int h    = bh & 31;
    const size_t hoff = (size_t)h * HDIM;

    // Q A-fragments (scaled by 1/sqrt(HD)=0.25): chunk c covers dims 4c..4c+3
    float qa0[4], qa1[4];
    {
        const size_t qro = ((size_t)(b * SEQ + q0 + c16)) * DMODEL + hoff;
        #pragma unroll
        for (int c = 0; c < 4; ++c) {
            int dim = c * 4 + 2 * half;
            qa0[c] = q[qro + dim]     * 0.25f;
            qa1[c] = q[qro + dim + 1] * 0.25f;
        }
    }

    v8f Oacc = {};
    float m[8], ssum[8];
    #pragma unroll
    for (int r = 0; r < 8; ++r) { m[r] = -1e30f; ssum[r] = 0.f; }

    for (int ti = 0; ti < 33; ++ti) {
        const int jt0 = q0 - 256 + ti * 16;

        // ---- S = (Q*scale) @ K^T via 4x v_wmma_f32_16x16x4_f32 ----
        v8f Sf = {};
        {
            int key  = jt0 + c16;
            int keyc = key < 0 ? 0 : (key >= SEQ ? SEQ - 1 : key);
            const size_t kro = ((size_t)(b * SEQ + keyc)) * DMODEL + hoff;
            #pragma unroll
            for (int c = 0; c < 4; ++c) {
                int dim = c * 4 + 2 * half;
                v2f av = { qa0[c], qa1[c] };
                v2f bv = { k[kro + dim], k[kro + dim + 1] };
                Sf = __builtin_amdgcn_wmma_f32_16x16x4_f32(false, av, false, bv,
                                                           (short)0, Sf, false, false);
            }
        }

        // ---- mask + online softmax (rows live across 16 lanes of a half) ----
        float parr[8];
        #pragma unroll
        for (int r = 0; r < 8; ++r) {
            int i = q0 + r + (half << 3);
            int j = jt0 + c16;
            bool ok = (j >= 0) & (j < SEQ) & (j - i <= WWIN) & (i - j <= WWIN);
            float sc = ok ? Sf[r] : -__builtin_inff();
            float rm = sc;
            rm = fmaxf(rm, __shfl_xor(rm, 1, 32));
            rm = fmaxf(rm, __shfl_xor(rm, 2, 32));
            rm = fmaxf(rm, __shfl_xor(rm, 4, 32));
            rm = fmaxf(rm, __shfl_xor(rm, 8, 32));
            float nm    = fmaxf(m[r], rm);
            float alpha = __expf(m[r] - nm);
            float p     = __expf(sc - nm);
            float rs = p;
            rs += __shfl_xor(rs, 1, 32);
            rs += __shfl_xor(rs, 2, 32);
            rs += __shfl_xor(rs, 4, 32);
            rs += __shfl_xor(rs, 8, 32);
            ssum[r] = ssum[r] * alpha + rs;
            Oacc[r] = Oacc[r] * alpha;
            m[r]    = nm;
            parr[r] = p;
        }

        // ---- transpose P (C-layout -> A-layout) through LDS ----
        __syncthreads();
        #pragma unroll
        for (int r = 0; r < 8; ++r)
            pT[(r + (half << 3)) * 17 + c16] = parr[r];
        __syncthreads();

        // ---- O += P @ V via 4x v_wmma_f32_16x16x4_f32 ----
        #pragma unroll
        for (int c = 0; c < 4; ++c) {
            int kk  = c * 4 + 2 * half;
            float pa0 = pT[c16 * 17 + kk];
            float pa1 = pT[c16 * 17 + kk + 1];
            int kr0 = jt0 + kk;     kr0 = kr0 < 0 ? 0 : (kr0 >= SEQ ? SEQ - 1 : kr0);
            int kr1 = jt0 + kk + 1; kr1 = kr1 < 0 ? 0 : (kr1 >= SEQ ? SEQ - 1 : kr1);
            v2f av = { pa0, pa1 };
            v2f bv = { v[((size_t)(b * SEQ + kr0)) * DMODEL + hoff + c16],
                       v[((size_t)(b * SEQ + kr1)) * DMODEL + hoff + c16] };
            Oacc = __builtin_amdgcn_wmma_f32_16x16x4_f32(false, av, false, bv,
                                                         (short)0, Oacc, false, false);
        }
    }

    #pragma unroll
    for (int r = 0; r < 8; ++r) {
        int M = r + (half << 3);
        o[((size_t)(b * SEQ + q0 + M)) * DMODEL + hoff + c16] = Oacc[r] / ssum[r];
    }
}

// ---------------------------------------------------------------------------
// Gather last DIFF rows per batch: xd[b*DIFF+i] = x[b*S + CTX + i]
// ---------------------------------------------------------------------------
__global__ __launch_bounds__(256)
void gather_tail_kernel(const float* __restrict__ x, float* __restrict__ xd)
{
    const int row = blockIdx.x;                  // 0..B*DIFF-1
    const int b   = row / DIFFN;
    const int i   = row % DIFFN;
    const size_t src = ((size_t)b * SEQ + CTXN + i) * DMODEL;
    xd[(size_t)row * DMODEL + threadIdx.x]       = x[src + threadIdx.x];
    xd[(size_t)row * DMODEL + threadIdx.x + 256] = x[src + threadIdx.x + 256];
}

// ---------------------------------------------------------------------------
extern "C" void kernel_launch(void* const* d_in, const int* in_sizes, int n_in,
                              void* d_out, int out_size, void* d_ws, size_t ws_size,
                              hipStream_t stream)
{
    (void)in_sizes; (void)n_in; (void)out_size; (void)ws_size;

    const int*   ids     = (const int*)  d_in[0];
    const float* noisy   = (const float*)d_in[1];
    const float* tvec    = (const float*)d_in[2];
    const float* ctx_emb = (const float*)d_in[3];
    const float* dproj_w = (const float*)d_in[4];
    const float* dproj_b = (const float*)d_in[5];
    const float* pos_emb = (const float*)d_in[6];
    const float* in_g    = (const float*)d_in[7];
    const float* in_b    = (const float*)d_in[8];
    const float* tw1     = (const float*)d_in[9];
    const float* tb1     = (const float*)d_in[10];
    const float* tw2     = (const float*)d_in[11];
    const float* tb2     = (const float*)d_in[12];
    const float* lf_pos  = (const float*)d_in[13];
    const float* lf_tok  = (const float*)d_in[14];
    const float* emb_g   = (const float*)d_in[15];
    const float* emb_b   = (const float*)d_in[16];
    const float* qw      = (const float*)d_in[17];
    const float* qb      = (const float*)d_in[18];
    const float* kw      = (const float*)d_in[19];
    const float* kb      = (const float*)d_in[20];
    const float* vw      = (const float*)d_in[21];
    const float* vb      = (const float*)d_in[22];
    const float* ow      = (const float*)d_in[23];
    const float* ob      = (const float*)d_in[24];
    const float* g1      = (const float*)d_in[25];
    const float* b1      = (const float*)d_in[26];
    const float* iw      = (const float*)d_in[27];
    const float* ib      = (const float*)d_in[28];
    const float* ow2     = (const float*)d_in[29];
    const float* ob2     = (const float*)d_in[30];
    const float* g2      = (const float*)d_in[31];
    const float* b2      = (const float*)d_in[32];
    const float* out_w   = (const float*)d_in[33];
    const float* out_b   = (const float*)d_in[34];
    float* out = (float*)d_out;

    const int ROWS = BB * SEQ;                   // 4000
    char* wsb = (char*)d_ws;
    size_t off = 0;
    auto alloc = [&](size_t bytes) -> float* {
        float* p = (float*)(wsb + off);
        off += (bytes + 255) & ~(size_t)255;
        return p;
    };
    float* x    = alloc((size_t)ROWS * DMODEL * 4);
    float* qb_  = alloc((size_t)ROWS * DMODEL * 4);
    float* kb_  = alloc((size_t)ROWS * DMODEL * 4);
    float* vb_  = alloc((size_t)ROWS * DMODEL * 4);
    float* ab_  = alloc((size_t)ROWS * DMODEL * 4);
    float* yb_  = alloc((size_t)ROWS * DMODEL * 4);
    float* ffb  = alloc((size_t)ROWS * FFD * 4);
    float* dbuf = alloc((size_t)BB * DIFFN * DMODEL * 4);
    float* tf   = alloc((size_t)BB * DMODEL * 4);
    float* th   = alloc((size_t)BB * FFD * 4);
    float* te   = alloc((size_t)BB * DMODEL * 4);
    float* xd   = alloc((size_t)BB * DIFFN * DMODEL * 4);

    auto gemm = [&](const float* A, const float* Wm, const float* bias, float* C,
                    int M, int K, int N, int act) {
        dim3 grid((N + 63) / 64, (M + 63) / 64);
        gemm_f16_kernel<<<grid, dim3(128), 0, stream>>>(A, Wm, bias, C, M, K, N, act);
    };

    // ---- embeddings ----
    gemm(noisy, dproj_w, dproj_b, dbuf, BB * DIFFN, VOCAB, DMODEL, 0);
    ln_kernel<<<BB * DIFFN, 256, 0, stream>>>(dbuf, nullptr, in_g, in_b, 1e-5f);

    tfreq_kernel<<<BB, 256, 0, stream>>>(tvec, tf);
    gemm(tf, tw1, tb1, th, BB, DMODEL, FFD, 1);      // GELU
    gemm(th, tw2, tb2, te, BB, FFD, DMODEL, 0);

    build_x_kernel<<<ROWS, 256, 0, stream>>>(ids, ctx_emb, dbuf, pos_emb, te,
                                             lf_pos, lf_tok, x);
    ln_kernel<<<ROWS, 256, 0, stream>>>(x, nullptr, emb_g, emb_b, 1e-12f);

    // ---- transformer layers ----
    for (int l = 0; l < NLAYER; ++l) {
        const size_t wdd = (size_t)l * DMODEL * DMODEL;
        const size_t wdf = (size_t)l * DMODEL * FFD;
        const size_t vd  = (size_t)l * DMODEL;
        const size_t vf  = (size_t)l * FFD;

        gemm(x, qw + wdd, qb + vd, qb_, ROWS, DMODEL, DMODEL, 0);
        gemm(x, kw + wdd, kb + vd, kb_, ROWS, DMODEL, DMODEL, 0);
        gemm(x, vw + wdd, vb + vd, vb_, ROWS, DMODEL, DMODEL, 0);

        band_attn_kernel<<<dim3(SEQ / 16, BB * NHEAD), 32, 0, stream>>>(qb_, kb_, vb_, ab_);

        gemm(ab_, ow + wdd, ob + vd, yb_, ROWS, DMODEL, DMODEL, 0);
        ln_kernel<<<ROWS, 256, 0, stream>>>(x, yb_, g1 + vd, b1 + vd, 1e-12f);

        gemm(x, iw + wdf, ib + vf, ffb, ROWS, DMODEL, FFD, 1);   // GELU
        gemm(ffb, ow2 + wdf, ob2 + vd, yb_, ROWS, FFD, DMODEL, 0);
        ln_kernel<<<ROWS, 256, 0, stream>>>(x, yb_, g2 + vd, b2 + vd, 1e-12f);
    }

    // ---- output head ----
    gather_tail_kernel<<<BB * DIFFN, 256, 0, stream>>>(x, xd);
    gemm(xd, out_w, out_b, out, BB * DIFFN, DMODEL, VOCAB, 0);
}
